// GCNSortPool_2568390443224
// MI455X (gfx1250) — matrix operand order repaired and checked by reading, not compiled, not run on hardware
//
#include <hip/hip_runtime.h>
#include <hip/hip_bf16.h>

typedef __attribute__((ext_vector_type(16))) __bf16 v16bf;
typedef __attribute__((ext_vector_type(8)))  float  v8f;

#define FEAT 256
#define HDIM 16
#define NGRAPH 512
#define TOPK 40

// ---------------- utility fills ----------------
__global__ void gsp_fill_f32(float* __restrict__ p, float v, long long n) {
    long long i = (long long)blockIdx.x * blockDim.x + threadIdx.x;
    if (i < n) p[i] = v;
}

// ---------------- degree / norm ----------------
__global__ void gsp_deg_count(const int* __restrict__ dst, float* __restrict__ deg, long long E) {
    long long e = (long long)blockIdx.x * blockDim.x + threadIdx.x;
    if (e < E) atomicAdd(&deg[dst[e]], 1.0f);
}

__global__ void gsp_rsqrt_inplace(float* __restrict__ d, int n) {
    int i = blockIdx.x * blockDim.x + threadIdx.x;
    if (i < n) d[i] = rsqrtf(d[i]);   // deg >= 1 always (self loops)
}

// ---------------- GEMM1: h1 = x @ W1  (N x 256) @ (256 x 16), bf16 WMMA ----------------
// One wave -> one 16x16 output tile. K loop: 8 steps of K=32.
__global__ void gsp_gemm1_wmma(const float* __restrict__ x, const float* __restrict__ W1,
                               float* __restrict__ h1, int N) {
    const int lane = threadIdx.x & 31;
    const int wave = threadIdx.x >> 5;
    const int tile = blockIdx.x * (blockDim.x >> 5) + wave;
    if (tile * 16 >= N) return;               // uniform per wave -> EXEC all-1 for WMMA

    const int row = lane & 15;                // M index for A
    const int hi  = lane >> 4;                // half-wave selector
    const int col = lane & 15;                // N index for B / C

    // Preload B (W1, K-major 256x16) for all 8 K-steps.
    // B 32x16 bf16 layout: lanes 0-15 hold K 0-7, 16-23 of column `col`; lanes 16-31 hold K 8-15, 24-31.
    v16bf Breg[8];
    #pragma unroll
    for (int kk = 0; kk < 8; ++kk) {
        const int kbase = kk * 32 + hi * 8;
        #pragma unroll
        for (int j = 0; j < 16; ++j)
            Breg[kk][j] = (__bf16)W1[(kbase + j + (j >= 8 ? 8 : 0)) * HDIM + col];
    }

    v8f acc = {};
    const float* xrow = x + (size_t)(tile * 16 + row) * FEAT;
    #pragma unroll
    for (int kk = 0; kk < 8; ++kk) {
        // A 16x32 bf16 layout: lanes 0-15: K 0-7 (elems 0..7), K 16-23 (elems 8..15);
        //                      lanes 16-31: K 8-15, K 24-31.
        const int k0 = kk * 32 + hi * 8;
        v16bf A;
        #pragma unroll
        for (int j = 0; j < 8; ++j) {
            A[j]     = (__bf16)xrow[k0 + j];
            A[8 + j] = (__bf16)xrow[k0 + 16 + j];
        }
        acc = __builtin_amdgcn_wmma_f32_16x16x32_bf16(false, A, false, Breg[kk],
                                                      (short)0, acc, false, false);
    }

    // C layout: lanes 0-15 -> N=col, M=r ; lanes 16-31 -> N=col, M=r+8
    const int mofs = hi * 8;
    float* out = h1 + (size_t)tile * 16 * HDIM;
    #pragma unroll
    for (int r = 0; r < 8; ++r)
        out[(mofs + r) * HDIM + col] = acc[r];
}

// ---------------- GEMM2: h2 = h @ W2  (N x 16) @ (16 x 16), single zero-padded WMMA ----------------
__global__ void gsp_gemm2_wmma(const float* __restrict__ hin, const float* __restrict__ W2,
                               float* __restrict__ hout, int N) {
    const int lane = threadIdx.x & 31;
    const int wave = threadIdx.x >> 5;
    const int tile = blockIdx.x * (blockDim.x >> 5) + wave;
    if (tile * 16 >= N) return;

    const int row = lane & 15;
    const int hi  = lane >> 4;
    const int col = lane & 15;

    v16bf B;                                   // K rows 16..31 are zero padding
    #pragma unroll
    for (int j = 0; j < 16; ++j)
        B[j] = (j >= 8) ? (__bf16)0.0f : (__bf16)W2[(hi * 8 + j) * HDIM + col];

    v16bf A;                                   // K 16..31 zero padding
    const float* hrow = hin + (size_t)(tile * 16 + row) * HDIM;
    #pragma unroll
    for (int j = 0; j < 8; ++j) {
        A[j]     = (__bf16)hrow[hi * 8 + j];
        A[8 + j] = (__bf16)0.0f;
    }

    v8f acc = {};
    acc = __builtin_amdgcn_wmma_f32_16x16x32_bf16(false, A, false, B, (short)0, acc, false, false);

    const int mofs = hi * 8;
    float* out = hout + (size_t)tile * 16 * HDIM;
    #pragma unroll
    for (int r = 0; r < 8; ++r)
        out[(mofs + r) * HDIM + col] = acc[r];
}

// ---------------- edge scatter: agg[dst] += dinv[src]*dinv[dst] * h[src]  ----------------
// one thread per (edge, channel); agg (12.8MB) and h (12.8MB) are L2-resident.
__global__ void gsp_scatter16(const int* __restrict__ src, const int* __restrict__ dst,
                              const float* __restrict__ dinv, const float* __restrict__ h,
                              float* __restrict__ agg, long long E) {
    long long t = (long long)blockIdx.x * blockDim.x + threadIdx.x;
    long long e = t >> 4;
    int c = (int)(t & 15);
    if (e >= E) return;
    int s = src[e], d = dst[e];
    float w = dinv[s] * dinv[d];
    atomicAdd(&agg[(size_t)d * HDIM + c], w * h[(size_t)s * HDIM + c]);
}

// ---------------- finalize: agg = relu(agg + dinv[n]^2 * hself + bias), optional key extract ----------------
__global__ void gsp_finalize(float* __restrict__ agg, const float* __restrict__ hself,
                             const float* __restrict__ dinv, const float* __restrict__ bias,
                             float* __restrict__ key, int N) {
    long long t = (long long)blockIdx.x * blockDim.x + threadIdx.x;
    if (t >= (long long)N * HDIM) return;
    int n = (int)(t >> 4), c = (int)(t & 15);
    float di = dinv[n];
    float v = agg[t] + di * di * hself[t] + bias[c];
    v = fmaxf(v, 0.0f);
    agg[t] = v;
    if (key && c == HDIM - 1) key[n] = v;
}

// ---------------- per-graph counts / starts ----------------
__global__ void gsp_count_batch(const int* __restrict__ batch, int* __restrict__ counts, int N) {
    int n = blockIdx.x * blockDim.x + threadIdx.x;
    if (n < N) atomicAdd(&counts[batch[n]], 1);
}

__global__ void gsp_scan512(const int* __restrict__ counts, int* __restrict__ starts) {
    if (threadIdx.x == 0 && blockIdx.x == 0) {
        int acc = 0;
        for (int g = 0; g < NGRAPH; ++g) { starts[g] = acc; acc += counts[g]; }
    }
}

// ---------------- stable descending rank within graph; write top-K rows ----------------
__global__ void gsp_rank_topk(const float* __restrict__ key, const int* __restrict__ batch,
                              const int* __restrict__ starts, const int* __restrict__ counts,
                              const float* __restrict__ h, float* __restrict__ pooled, int N) {
    int n = blockIdx.x * blockDim.x + threadIdx.x;
    if (n >= N) return;
    int g = batch[n];
    int s = starts[g], cnt = counts[g];
    float myk = key[n];
    int rank = 0;
    for (int m = s; m < s + cnt; ++m) {
        float k2 = key[m];
        rank += (k2 > myk) || (k2 == myk && m < n);   // lexsort stable desc tie-break
    }
    if (rank < TOPK) {
        float* dp = pooled + ((size_t)g * TOPK + rank) * HDIM;
        const float* sp = h + (size_t)n * HDIM;
        #pragma unroll
        for (int c = 0; c < HDIM; ++c) dp[c] = sp[c];
    }
}

// ---------------- final FC: out[g] = pooled[g] . fc_w + fc_b ----------------
__global__ void gsp_fc(const float* __restrict__ pooled, const float* __restrict__ fc_w,
                       const float* __restrict__ fc_b, float* __restrict__ out) {
    int g = blockIdx.x * blockDim.x + threadIdx.x;
    if (g >= NGRAPH) return;
    float acc = fc_b[0];
    const float* p = pooled + (size_t)g * (TOPK * HDIM);
    #pragma unroll 4
    for (int i = 0; i < TOPK * HDIM; ++i) acc += p[i] * fc_w[i];
    out[g] = acc;
}

extern "C" void kernel_launch(void* const* d_in, const int* in_sizes, int n_in,
                              void* d_out, int out_size, void* d_ws, size_t ws_size,
                              hipStream_t stream) {
    const float* x     = (const float*)d_in[0];
    const int*   eidx  = (const int*)d_in[1];
    const int*   batch = (const int*)d_in[2];
    // d_in[3] = edge_weight (unused by reference forward)
    const float* W1    = (const float*)d_in[4];
    const float* b1    = (const float*)d_in[5];
    const float* W2    = (const float*)d_in[6];
    const float* b2    = (const float*)d_in[7];
    const float* fc_w  = (const float*)d_in[8];
    const float* fc_b  = (const float*)d_in[9];
    float* out = (float*)d_out;

    const int       N = in_sizes[0] / FEAT;         // 200000
    const long long E = (long long)in_sizes[1] / 2; // 3200000
    const int* src = eidx;
    const int* dst = eidx + E;

    // workspace carving (all sizes in bytes)
    char* ws = (char*)d_ws;
    float* dinv   = (float*)ws;                                          // N
    float* bufA   = (float*)(ws + (size_t)N * 4);                        // N*16 (h1, then h2)
    float* bufB   = (float*)(ws + (size_t)N * 4 + (size_t)N * 64);       // N*16 (agg/relu, reused)
    float* key    = (float*)(ws + (size_t)N * 4 + (size_t)N * 128);      // N
    int*   counts = (int*)  (ws + (size_t)N * 4 + (size_t)N * 128 + (size_t)N * 4);
    int*   starts = counts + NGRAPH;
    float* pooled = (float*)(starts + NGRAPH);                           // G*K*16

    const int TB = 256;
    const long long NH = (long long)N * HDIM;
    const long long ET = E * HDIM;
    const int tiles  = N / 16;
    const int gBlk   = (tiles + (TB / 32) - 1) / (TB / 32);
    const long long pooledN = (long long)NGRAPH * TOPK * HDIM;

    // 1) degree (self-loop => init 1.0) -> dinv
    gsp_fill_f32<<<(N + TB - 1) / TB, TB, 0, stream>>>(dinv, 1.0f, N);
    gsp_deg_count<<<(int)((E + TB - 1) / TB), TB, 0, stream>>>(dst, dinv, E);
    gsp_rsqrt_inplace<<<(N + TB - 1) / TB, TB, 0, stream>>>(dinv, N);

    // 2) layer 1: GEMM (WMMA) -> scatter -> relu finalize
    gsp_gemm1_wmma<<<gBlk, TB, 0, stream>>>(x, W1, bufA, N);
    gsp_fill_f32<<<(int)((NH + TB - 1) / TB), TB, 0, stream>>>(bufB, 0.0f, NH);
    gsp_scatter16<<<(int)((ET + TB - 1) / TB), TB, 0, stream>>>(src, dst, dinv, bufA, bufB, E);
    gsp_finalize<<<(int)((NH + TB - 1) / TB), TB, 0, stream>>>(bufB, bufA, dinv, b1, nullptr, N);

    // 3) layer 2: GEMM (WMMA) -> scatter -> relu finalize (+ sort key = channel 15)
    gsp_gemm2_wmma<<<gBlk, TB, 0, stream>>>(bufB, W2, bufA, N);
    gsp_fill_f32<<<(int)((NH + TB - 1) / TB), TB, 0, stream>>>(bufB, 0.0f, NH);
    gsp_scatter16<<<(int)((ET + TB - 1) / TB), TB, 0, stream>>>(src, dst, dinv, bufA, bufB, E);
    gsp_finalize<<<(int)((NH + TB - 1) / TB), TB, 0, stream>>>(bufB, bufA, dinv, b2, key, N);

    // 4) per-graph segmentation (batch is sorted -> contiguous ranges)
    gsp_fill_f32<<<(NGRAPH + TB - 1) / TB, TB, 0, stream>>>((float*)counts, 0.0f, NGRAPH);
    gsp_count_batch<<<(N + TB - 1) / TB, TB, 0, stream>>>(batch, counts, N);
    gsp_scan512<<<1, 1, 0, stream>>>(counts, starts);

    // 5) top-K selection via stable rank + FC head
    gsp_fill_f32<<<(int)((pooledN + TB - 1) / TB), TB, 0, stream>>>(pooled, 0.0f, pooledN);
    gsp_rank_topk<<<(N + TB - 1) / TB, TB, 0, stream>>>(key, batch, starts, counts, bufB, pooled, N);
    gsp_fc<<<(NGRAPH + 63) / 64, 64, 0, stream>>>(pooled, fc_w, fc_b, out);
}
